// GIN_NET_1039382085872
// MI455X (gfx1250) — matrix operand order repaired and checked by reading, not compiled, not run on hardware
//
#include <hip/hip_runtime.h>
#include <hip/hip_bf16.h>

#define N_NODES 100000
#define N_EDGES 1600000
#define IN_DIM 64
#define HID_DIM 128
#define OUT_DIM 128
#define N_GRAPHS 256
#define BN_EPS 1e-5f

typedef __attribute__((ext_vector_type(16))) __bf16 v16bf;
typedef __attribute__((ext_vector_type(8)))  __bf16 v8bf;
typedef __attribute__((ext_vector_type(8)))  float  v8f;

// ---------------- workspace layout (bytes) ----------------
// agg (f32, 100000x64)      @ 0            25,600,000
// h0  (bf16,100000x64)      @ 25,600,000   12,800,000
// h1  (bf16,100000x128)     @ 38,400,000   25,600,000
// h2  (bf16,100000x128)     @ 64,000,000   25,600,000
// h3  (f32, 100000x128)     @ 0            51,200,000   (aliases agg/h0/h1, dead by then)
// W1t (bf16,128x64)         @ 89,600,000       16,384
// W2t (bf16,128x128)        @ 89,616,384       32,768
// W3t (bf16,128x128)        @ 89,649,152       32,768
// stats: sums[128] sumsq[128] scale[128] shift[128] f32 @ 89,681,920  2,048
#define AGG_OFF   0ULL
#define H0_OFF    25600000ULL
#define H1_OFF    38400000ULL
#define H2_OFF    64000000ULL
#define H3_OFF    0ULL
#define W1T_OFF   89600000ULL
#define W2T_OFF   89616384ULL
#define W3T_OFF   89649152ULL
#define STATS_OFF 89681920ULL

__global__ void zero_f32(float* __restrict__ p, long long n) {
    long long i = (long long)blockIdx.x * blockDim.x + threadIdx.x;
    long long stride = (long long)gridDim.x * blockDim.x;
    for (; i < n; i += stride) p[i] = 0.0f;
}

// Transpose + convert the three weight matrices to bf16 [N x K] row-major.
__global__ void prep_weights(const float* __restrict__ W1, const float* __restrict__ W2,
                             const float* __restrict__ W3, __bf16* __restrict__ W1t,
                             __bf16* __restrict__ W2t, __bf16* __restrict__ W3t) {
    int idx = blockIdx.x * blockDim.x + threadIdx.x;
    if (idx < 64 * 128) {                       // W1[k*128+n] -> W1t[n*64+k]
        int n = idx & 127, k = idx >> 7;
        W1t[n * 64 + k] = (__bf16)W1[idx];
    } else if (idx < 64 * 128 + 128 * 128) {    // W2
        int j = idx - 64 * 128;
        int n = j & 127, k = j >> 7;
        W2t[n * 128 + k] = (__bf16)W2[j];
    } else if (idx < 64 * 128 + 2 * 128 * 128) {// W3
        int j = idx - (64 * 128 + 128 * 128);
        int n = j & 127, k = j >> 7;
        W3t[n * 128 + k] = (__bf16)W3[j];
    }
}

// Edge scatter-add: agg[dst] += x[src].  One thread per (edge, float4-chunk).
__global__ void scatter_add(const float* __restrict__ x, const long long* __restrict__ ei,
                            float* __restrict__ agg) {
    int idx = blockIdx.x * blockDim.x + threadIdx.x;
    if (idx >= N_EDGES * 16) return;
    int e = idx >> 4;
    int q = idx & 15;
    long long s = ei[e];
    long long d = ei[N_EDGES + e];
    const float4 v = *(const float4*)(x + s * IN_DIM + q * 4);
    float* o = agg + d * IN_DIM + q * 4;
    atomicAdd(o + 0, v.x);
    atomicAdd(o + 1, v.y);
    atomicAdd(o + 2, v.z);
    atomicAdd(o + 3, v.w);
}

// h0 = bf16(x + agg)
__global__ void add_cvt(const float* __restrict__ x, const float* __restrict__ agg,
                        __bf16* __restrict__ h0) {
    long long i = (long long)blockIdx.x * blockDim.x + threadIdx.x;
    if (i < (long long)N_NODES * IN_DIM) h0[i] = (__bf16)(x[i] + agg[i]);
}

// GEMM + bias + ReLU via v_wmma_f32_16x16x32_bf16.
// A: [M x K] bf16 row-major.  Bt: [128 x K] bf16 row-major (i.e. W transposed).
// One wave = one 16x16 output tile; block = 16 rows x 128 cols (8 waves).
template <int K, bool OUT_F32>
__global__ __launch_bounds__(256) void gemm_bias_relu(const __bf16* __restrict__ A,
                                                      const __bf16* __restrict__ Bt,
                                                      const float* __restrict__ bias,
                                                      __bf16* __restrict__ Obf,
                                                      float* __restrict__ Of32) {
    const int lane  = threadIdx.x & 31;
    const int wave  = threadIdx.x >> 5;   // N-tile 0..7
    const int mtile = blockIdx.x;         // 0..6249
    const int half  = lane >> 4;          // 0/1
    const int l15   = lane & 15;
    const long long m = (long long)mtile * 16 + l15;  // A row for this lane
    const int       n = wave * 16 + l15;              // output column for this lane

    v8f c = {};
    #pragma unroll
    for (int kt = 0; kt < K; kt += 32) {
        // A fragment: lane holds row m, 16 contiguous K starting at kt + 16*half
        const int ka = kt + half * 16;
        v16bf a = *(const v16bf*)(A + m * K + ka);
        // B fragment: lane holds column n, 16 contiguous K starting at kt + 16*half
        const int kb = kt + half * 16;
        v16bf b = *(const v16bf*)(Bt + (long long)n * K + kb);
        c = __builtin_amdgcn_wmma_f32_16x16x32_bf16(false, a, false, b, (short)0, c,
                                                    false, false);
    }
    const float bn = bias[n];
    #pragma unroll
    for (int v = 0; v < 8; ++v) {
        long long row = (long long)mtile * 16 + v + half * 8;
        float r = c[v] + bn;
        r = r > 0.0f ? r : 0.0f;
        if (OUT_F32) Of32[row * 128 + n] = r;
        else         Obf [row * 128 + n] = (__bf16)r;
    }
}

// Per-channel sum / sum-of-squares over h3 (100000 x 128), 512 rows per block.
__global__ void bn_stats(const float* __restrict__ h3, float* __restrict__ sums,
                         float* __restrict__ sumsq) {
    const int c = threadIdx.x;                 // 0..127
    long long r0 = (long long)blockIdx.x * 512;
    float s = 0.0f, q = 0.0f;
    for (int r = 0; r < 512; ++r) {
        long long row = r0 + r;
        if (row < N_NODES) {
            float v = h3[row * 128 + c];
            s += v;
            q += v * v;
        }
    }
    atomicAdd(&sums[c], s);
    atomicAdd(&sumsq[c], q);
}

// scale/shift from batch statistics (biased variance).
__global__ void bn_finalize(const float* __restrict__ sums, const float* __restrict__ sumsq,
                            const float* __restrict__ gamma, const float* __restrict__ beta,
                            float* __restrict__ scale, float* __restrict__ shift) {
    int c = threadIdx.x;
    float mean = sums[c] * (1.0f / (float)N_NODES);
    float var  = sumsq[c] * (1.0f / (float)N_NODES) - mean * mean;
    float sc   = gamma[c] * rsqrtf(var + BN_EPS);
    scale[c] = sc;
    shift[c] = beta[c] - mean * sc;
}

// out[batch[i]] += h3[i]*scale + shift
__global__ void pool(const float* __restrict__ h3, const long long* __restrict__ batch,
                     const float* __restrict__ scale, const float* __restrict__ shift,
                     float* __restrict__ out) {
    long long idx = (long long)blockIdx.x * blockDim.x + threadIdx.x;
    if (idx >= (long long)N_NODES * 128) return;
    int c = (int)(idx & 127);
    long long i = idx >> 7;
    long long g = batch[i];
    float v = h3[idx] * scale[c] + shift[c];
    atomicAdd(&out[g * 128 + c], v);
}

extern "C" void kernel_launch(void* const* d_in, const int* in_sizes, int n_in,
                              void* d_out, int out_size, void* d_ws, size_t ws_size,
                              hipStream_t stream) {
    (void)in_sizes; (void)n_in; (void)out_size; (void)ws_size;
    const float*      x     = (const float*)d_in[0];
    const long long*  ei    = (const long long*)d_in[1];
    const long long*  batch = (const long long*)d_in[2];
    const float*      W1    = (const float*)d_in[3];
    const float*      b1    = (const float*)d_in[4];
    const float*      W2    = (const float*)d_in[5];
    const float*      b2    = (const float*)d_in[6];
    const float*      W3    = (const float*)d_in[7];
    const float*      b3    = (const float*)d_in[8];
    const float*      gamma = (const float*)d_in[9];
    const float*      beta  = (const float*)d_in[10];
    float* out = (float*)d_out;
    char*  ws  = (char*)d_ws;

    float*  agg   = (float*)(ws + AGG_OFF);
    __bf16* h0    = (__bf16*)(ws + H0_OFF);
    __bf16* h1    = (__bf16*)(ws + H1_OFF);
    __bf16* h2    = (__bf16*)(ws + H2_OFF);
    float*  h3    = (float*)(ws + H3_OFF);
    __bf16* W1t   = (__bf16*)(ws + W1T_OFF);
    __bf16* W2t   = (__bf16*)(ws + W2T_OFF);
    __bf16* W3t   = (__bf16*)(ws + W3T_OFF);
    float*  sums  = (float*)(ws + STATS_OFF);
    float*  sumsq = sums + 128;
    float*  scale = sums + 256;
    float*  shift = sums + 384;

    // init: zero agg, BN accumulators, output
    zero_f32<<<2048, 256, 0, stream>>>(agg, (long long)N_NODES * IN_DIM);
    zero_f32<<<2, 256, 0, stream>>>(sums, 256);
    zero_f32<<<128, 256, 0, stream>>>(out, (long long)N_GRAPHS * OUT_DIM);

    // weight transpose + bf16 convert
    prep_weights<<<(64 * 128 + 2 * 128 * 128 + 255) / 256, 256, 0, stream>>>(
        W1, W2, W3, W1t, W2t, W3t);

    // edge aggregation
    scatter_add<<<(N_EDGES * 16 + 255) / 256, 256, 0, stream>>>(x, ei, agg);

    // h0 = bf16(x + agg)
    add_cvt<<<((long long)N_NODES * IN_DIM + 255) / 256, 256, 0, stream>>>(x, agg, h0);

    // MLP: three WMMA GEMMs with fused bias+ReLU
    gemm_bias_relu<64, false><<<N_NODES / 16, 256, 0, stream>>>(h0, W1t, b1, h1, nullptr);
    gemm_bias_relu<128, false><<<N_NODES / 16, 256, 0, stream>>>(h1, W2t, b2, h2, nullptr);
    gemm_bias_relu<128, true><<<N_NODES / 16, 256, 0, stream>>>(h2, W3t, b3, nullptr, h3);

    // BatchNorm statistics + affine fold
    bn_stats<<<(N_NODES + 511) / 512, 128, 0, stream>>>(h3, sums, sumsq);
    bn_finalize<<<1, 128, 0, stream>>>(sums, sumsq, gamma, beta, scale, shift);

    // normalized global_add_pool
    pool<<<((long long)N_NODES * 128 + 255) / 256, 256, 0, stream>>>(h3, batch, scale,
                                                                     shift, out);
}
